// GraphSAGE_1529008357611
// MI455X (gfx1250) — compile-verified, matrix operand order checked by reading, not run on hardware
//
#include <hip/hip_runtime.h>
#include <hip/hip_bf16.h>

#define N_ROI    116
#define HIDDEN   64
#define N_NODES  118784LL       // 1024 graphs * 116 ROIs
#define N_EDGES  1900544LL
#define BATCH    1024
#define D_FLAT   7424           // 116*64
#define D1       3712           // 116*32

typedef __attribute__((ext_vector_type(16))) __bf16 v16bf;
typedef __attribute__((ext_vector_type(8)))  float  v8f;

__device__ __forceinline__ float mish_f(float v) {
    // mish(x) = x * tanh(log(1 + exp(x)))
    float sp = log1pf(__expf(v));
    return v * tanhf(sp);
}

__device__ __forceinline__ v8f wmma_bf16(v16bf a, v16bf b, v8f c) {
    return __builtin_amdgcn_wmma_f32_16x16x32_bf16(false, a, false, b,
                                                   (short)0, c, false, false);
}

// ---------------------------------------------------------------------------
// Operand packing for v_wmma_f32_16x16x32_bf16 (wave32).
// Packed layout = the WMMA register file layout, so the GEMM kernels read
// contiguous 32 B per lane per K-step (two global_load_b128):
//   pack[tile][kt][lane][e],  lane: rc = lane&15, kb = (lane>>4)*8,
//   half e in [0,16): k = kt*32 + kb + (e < 8 ? e : e + 8)
// A tile: lane holds row (mt*16+rc).  B tile: lane holds col (nt*16+rc) of
// the row-major [K][N] weight. K padded to KT*32 with zeros at pack time.
// ---------------------------------------------------------------------------
__global__ void pack_a_bf16(const float* __restrict__ A,
                            const float* __restrict__ cnt,   // optional mean-divide
                            v16bf* __restrict__ Out, int Kdim, int KT) {
    int lane = threadIdx.x & 31;
    int rc   = lane & 15;
    int kb   = (lane >> 4) * 8;
    int mt   = blockIdx.x;
    int kt   = blockIdx.y;
    int row  = mt * 16 + rc;

    float scale = 1.0f;
    if (cnt) scale = 1.0f / fmaxf(cnt[row], 1.0f);

    const float* Ar = A + (long long)row * Kdim;
    int k0 = kt * 32;
    v16bf v;
#pragma unroll
    for (int e = 0; e < 16; ++e) {
        int k  = k0 + kb + (e < 8 ? e : e + 8);
        int kc = (k < Kdim) ? k : 0;           // clamped addr: no exec branching
        float f = Ar[kc] * scale;
        v[e] = (__bf16)((k < Kdim) ? f : 0.0f);
    }
    Out[((size_t)mt * KT + kt) * 32 + lane] = v;
}

__global__ void pack_w_bf16(const float* __restrict__ W,    // [K][N] row-major
                            v16bf* __restrict__ Out, int Kdim, int Nw, int KT) {
    int lane = threadIdx.x & 31;
    int rc   = lane & 15;
    int kb   = (lane >> 4) * 8;
    int nt   = blockIdx.x;
    int kt   = blockIdx.y;
    int col  = nt * 16 + rc;
    int colc = (col < Nw) ? col : 0;

    int k0 = kt * 32;
    v16bf v;
#pragma unroll
    for (int e = 0; e < 16; ++e) {
        int k  = k0 + kb + (e < 8 ? e : e + 8);
        int kc = (k < Kdim) ? k : 0;
        float f = W[(long long)kc * Nw + colc];
        v[e] = (__bf16)((k < Kdim && col < Nw) ? f : 0.0f);
    }
    Out[((size_t)nt * KT + kt) * 32 + lane] = v;
}

// ---------------------------------------------------------------------------
// Utility kernels
// ---------------------------------------------------------------------------
__global__ void zero_f32(float* __restrict__ p, long long n) {
    long long stride = (long long)gridDim.x * blockDim.x;
    for (long long i = (long long)blockIdx.x * blockDim.x + threadIdx.x; i < n; i += stride)
        p[i] = 0.0f;
}

__global__ void count_edges(const long long* __restrict__ dst, float* __restrict__ cnt,
                            long long E) {
    long long stride = (long long)gridDim.x * blockDim.x;
    for (long long e = (long long)blockIdx.x * blockDim.x + threadIdx.x; e < E; e += stride)
        atomicAdd(&cnt[dst[e]], 1.0f);
}

// scatter-add of feature rows: sum[dst] += x[src]  (55 MB accumulator -> L2 atomics)
__global__ void scatter_feat(const float* __restrict__ x,
                             const long long* __restrict__ src,
                             const long long* __restrict__ dst,
                             float* __restrict__ sum, int F, long long E) {
    long long total = E * (long long)F;
    long long stride = (long long)gridDim.x * blockDim.x;
    for (long long i = (long long)blockIdx.x * blockDim.x + threadIdx.x; i < total; i += stride) {
        long long e = i / F;
        int f = (int)(i - e * (long long)F);
        atomicAdd(&sum[dst[e] * F + f], x[src[e] * F + f]);
    }
}

// ---------------------------------------------------------------------------
// Fused SAGEConv node update on packed operands:
//   out = mish( mean @ Wl + bl + X @ Wr )
// One wave per 16x16 tile; inner loop: 4x v16bf loads (8x b128) + 2x WMMA.
// ---------------------------------------------------------------------------
__global__ void sage_gemm_packed(const v16bf* __restrict__ Am,  // packed mean
                                 const v16bf* __restrict__ Wl,
                                 const v16bf* __restrict__ Ax,  // packed x
                                 const v16bf* __restrict__ Wr,
                                 const float* __restrict__ bias,
                                 float* __restrict__ out,
                                 int KT, int Ndim) {
    int lane = threadIdx.x & 31;
    int mt = blockIdx.x, nt = blockIdx.y;
    const v16bf* am = Am + (size_t)mt * KT * 32 + lane;
    const v16bf* wl = Wl + (size_t)nt * KT * 32 + lane;
    const v16bf* ax = Ax + (size_t)mt * KT * 32 + lane;
    const v16bf* wr = Wr + (size_t)nt * KT * 32 + lane;

    v8f acc = {};
#pragma unroll 4
    for (int kt = 0; kt < KT; ++kt) {
        acc = wmma_bf16(am[kt * 32], wl[kt * 32], acc);
        acc = wmma_bf16(ax[kt * 32], wr[kt * 32], acc);
    }

    int col = nt * 16 + (lane & 15);
    if (col < Ndim) {
        float bv = bias[col];
#pragma unroll
        for (int v = 0; v < 8; ++v) {
            int r = mt * 16 + v + ((lane >> 4) << 3);
            out[(long long)r * Ndim + col] = mish_f(acc[v] + bv);
        }
    }
}

// ---------------------------------------------------------------------------
// Dense head GEMM on packed operands: out[M,N] = A @ W + bias (no activation).
// 4 waves/block, each owning one 16-wide N tile. KT = 232.
// ---------------------------------------------------------------------------
__global__ void lin_gemm_packed(const v16bf* __restrict__ A,
                                const v16bf* __restrict__ W,
                                const float* __restrict__ bias,
                                float* __restrict__ out,
                                int KT, int Ndim) {
    int wave = threadIdx.x >> 5;
    int lane = threadIdx.x & 31;
    int mt = blockIdx.x;
    int nt = blockIdx.y * 4 + wave;

    const v16bf* ap = A + (size_t)mt * KT * 32 + lane;
    const v16bf* wp = W + (size_t)nt * KT * 32 + lane;

    v8f acc = {};
    for (int kt = 0; kt < KT; ++kt) {
        if (kt + 1 < KT)  // uniform branch: prefetch next weight K-step
            __builtin_prefetch(wp + (kt + 1) * 32, 0, 1);
        acc = wmma_bf16(ap[kt * 32], wp[kt * 32], acc);
    }

    int col = nt * 16 + (lane & 15);          // Ndim = 3712, always in range
    float bv = bias[col];
#pragma unroll
    for (int v = 0; v < 8; ++v) {
        int r = mt * 16 + v + ((lane >> 4) << 3);
        out[(long long)r * Ndim + col] = acc[v] + bv;
    }
}

// ---------------------------------------------------------------------------
// Per-row layernorm + mish + [D1 -> 2] projection. One block per graph row.
// ---------------------------------------------------------------------------
__global__ void head_ln_mish_lin2(const float* __restrict__ y,
                                  const float* __restrict__ g,
                                  const float* __restrict__ bln,
                                  const float* __restrict__ w2,
                                  const float* __restrict__ b2,
                                  float* __restrict__ out, int D) {
    __shared__ float s0[256];
    __shared__ float s1[256];
    int tid = threadIdx.x;
    const float* yr = y + (long long)blockIdx.x * D;

    float sum = 0.0f, sq = 0.0f;
    for (int j = tid; j < D; j += blockDim.x) {
        float v = yr[j];
        sum += v;
        sq  += v * v;
    }
    s0[tid] = sum; s1[tid] = sq;
    __syncthreads();
    for (int off = 128; off > 0; off >>= 1) {
        if (tid < off) { s0[tid] += s0[tid + off]; s1[tid] += s1[tid + off]; }
        __syncthreads();
    }
    float mu   = s0[0] / (float)D;
    float var  = s1[0] / (float)D - mu * mu;
    float rstd = rsqrtf(var + 1e-5f);
    __syncthreads();

    float a0 = 0.0f, a1 = 0.0f;
    for (int j = tid; j < D; j += blockDim.x) {
        float h = (yr[j] - mu) * rstd * g[j] + bln[j];
        float m = mish_f(h);
        a0 += m * w2[2 * j + 0];
        a1 += m * w2[2 * j + 1];
    }
    s0[tid] = a0; s1[tid] = a1;
    __syncthreads();
    for (int off = 128; off > 0; off >>= 1) {
        if (tid < off) { s0[tid] += s0[tid + off]; s1[tid] += s1[tid + off]; }
        __syncthreads();
    }
    if (tid == 0) {
        out[blockIdx.x * 2 + 0] = s0[0] + b2[0];
        out[blockIdx.x * 2 + 1] = s1[0] + b2[1];
    }
}

// ---------------------------------------------------------------------------
extern "C" void kernel_launch(void* const* d_in, const int* in_sizes, int n_in,
                              void* d_out, int out_size, void* d_ws, size_t ws_size,
                              hipStream_t stream) {
    const float*     x      = (const float*)d_in[0];
    const long long* ei     = (const long long*)d_in[1];
    const float*     w1_l   = (const float*)d_in[2];
    const float*     b1_l   = (const float*)d_in[3];
    const float*     w1_r   = (const float*)d_in[4];
    const float*     w2_l   = (const float*)d_in[5];
    const float*     b2_l   = (const float*)d_in[6];
    const float*     w2_r   = (const float*)d_in[7];
    const float*     w_lin1 = (const float*)d_in[8];
    const float*     b_lin1 = (const float*)d_in[9];
    const float*     ln_g   = (const float*)d_in[10];
    const float*     ln_b   = (const float*)d_in[11];
    const float*     w_lin2 = (const float*)d_in[12];
    const float*     b_lin2 = (const float*)d_in[13];

    const long long* src = ei;             // edge_index[0]
    const long long* dst = ei + N_EDGES;   // edge_index[1]

    // ----- workspace layout (256B-aligned regions) -----
    const size_t SZ_SUM  = (size_t)N_NODES * N_ROI;      // 13,778,944 f32
    const size_t SZ_CNT  = (size_t)N_NODES;              //    118,784 f32
    const size_t SZ_H1   = (size_t)N_NODES * N_ROI;
    const size_t SZ_H2   = (size_t)N_NODES * HIDDEN;     //  7,602,176 f32
    const size_t SZ_Y1   = (size_t)BATCH * D1;           //  3,801,088 f32
    const size_t SZ_PA   = (size_t)(N_NODES / 16) * 4 * 32; // 950,272 v16bf (30.4MB)

    float* f      = (float*)d_ws;
    float* sumbuf = f;
    float* cntbuf = sumbuf + SZ_SUM;
    float* h1     = cntbuf + SZ_CNT;
    float* h2     = h1 + SZ_H1;
    float* y1     = h2 + SZ_H2;
    v16bf* packA0 = (v16bf*)(y1 + SZ_Y1);                // 30.4 MB
    v16bf* packA1 = packA0 + SZ_PA;                      // 30.4 MB
    v16bf* wp_l   = packA1 + SZ_PA;                      // small weight packs
    v16bf* wp_r   = wp_l + 8 * 4 * 32;
    // Big packed w_lin1 (55.1 MB) overlays sumbuf: dead by lin1 time.
    v16bf* packW  = (v16bf*)sumbuf;                      // 232*232*32 v16bf

    const int MT = (int)(N_NODES / 16);                  // 7424 node tiles

    // ---- layer 1: aggregate ----
    zero_f32<<<2048, 256, 0, stream>>>(sumbuf, (long long)(SZ_SUM + SZ_CNT));
    count_edges<<<2048, 256, 0, stream>>>(dst, cntbuf, N_EDGES);
    scatter_feat<<<4096, 256, 0, stream>>>(x, src, dst, sumbuf, N_ROI, N_EDGES);

    // ---- layer 1: pack operands + WMMA node update ----
    pack_a_bf16<<<dim3(MT, 4), 32, 0, stream>>>(sumbuf, cntbuf, packA0, N_ROI, 4);
    pack_a_bf16<<<dim3(MT, 4), 32, 0, stream>>>(x, nullptr, packA1, N_ROI, 4);
    pack_w_bf16<<<dim3(8, 4), 32, 0, stream>>>(w1_l, wp_l, N_ROI, N_ROI, 4);
    pack_w_bf16<<<dim3(8, 4), 32, 0, stream>>>(w1_r, wp_r, N_ROI, N_ROI, 4);
    sage_gemm_packed<<<dim3(MT, 8), 32, 0, stream>>>(packA0, wp_l, packA1, wp_r,
                                                     b1_l, h1, 4, N_ROI);

    // ---- layer 2: aggregate h1, pack, WMMA ----
    zero_f32<<<2048, 256, 0, stream>>>(sumbuf, (long long)SZ_SUM);
    scatter_feat<<<4096, 256, 0, stream>>>(h1, src, dst, sumbuf, N_ROI, N_EDGES);
    pack_a_bf16<<<dim3(MT, 4), 32, 0, stream>>>(sumbuf, cntbuf, packA0, N_ROI, 4);
    pack_a_bf16<<<dim3(MT, 4), 32, 0, stream>>>(h1, nullptr, packA1, N_ROI, 4);
    pack_w_bf16<<<dim3(4, 4), 32, 0, stream>>>(w2_l, wp_l, N_ROI, HIDDEN, 4);
    pack_w_bf16<<<dim3(4, 4), 32, 0, stream>>>(w2_r, wp_r, N_ROI, HIDDEN, 4);
    sage_gemm_packed<<<dim3(MT, 4), 32, 0, stream>>>(packA0, wp_l, packA1, wp_r,
                                                     b2_l, h2, 4, HIDDEN);

    // ---- head GEMM: [1024,7424] @ [7424,3712] + b ----
    pack_a_bf16<<<dim3(BATCH / 16, D_FLAT / 32), 32, 0, stream>>>(h2, nullptr, packA0,
                                                                  D_FLAT, D_FLAT / 32);
    pack_w_bf16<<<dim3(D1 / 16, D_FLAT / 32), 32, 0, stream>>>(w_lin1, packW,
                                                               D_FLAT, D1, D_FLAT / 32);
    lin_gemm_packed<<<dim3(BATCH / 16, D1 / 64), 128, 0, stream>>>(packA0, packW,
                                                                   b_lin1, y1,
                                                                   D_FLAT / 32, D1);

    // ---- layernorm + mish + [3712 -> 2] ----
    head_ln_mish_lin2<<<BATCH, 256, 0, stream>>>(y1, ln_g, ln_b, w_lin2, b_lin2,
                                                 (float*)d_out, D1);
}
